// BasicBlock6_4947802325429
// MI455X (gfx1250) — matrix operand order repaired and checked by reading, not compiled, not run on hardware
//
#include <hip/hip_runtime.h>
#include <hip/hip_bf16.h>

typedef __attribute__((ext_vector_type(16))) _Float16 v16h;
typedef __attribute__((ext_vector_type(8)))  float    v8f;

#define BB   8
#define CC   128
#define HH   64
#define WW   64
#define KKT  9          // 3x3 taps
#define KDIM 1152       // Ci * 9
#define NKC  36         // KDIM / 32
#define AROW 1160       // padded row (halves) for A in LDS

// ---------------------------------------------------------------------------
// Pack deform-conv weights (Co=128, Ci=128, 3,3) fp32 -> f16 B-fragment layout:
// Bpack[nt][kc][lane][e], e=0..15 halves. lane: N = nt*16 + (lane&15),
// K = kc*32 + (lane>>4)*8 + (e&7) + (e>>3)*16  (matches 16-bit B 32x16 fragment striping).
// K index maps to weight as kg = ci*9 + kk (einsum 'oik' ordering).
// ---------------------------------------------------------------------------
__global__ void pack_w_kernel(const float* __restrict__ w, _Float16* __restrict__ bp) {
    int idx = blockIdx.x * 256 + threadIdx.x;
    if (idx >= 8 * NKC * 32 * 16) return;
    int e    = idx & 15;
    int lane = (idx >> 4) & 31;
    int kc   = (idx >> 9) % NKC;
    int nt   = idx / (16 * 32 * NKC);
    int n  = nt * 16 + (lane & 15);
    int kg = kc * 32 + (lane >> 4) * 8 + (e & 7) + (e >> 3) * 16;
    int ci = kg / KKT;
    int kk = kg % KKT;
    bp[idx] = (_Float16)w[(n * CC + ci) * KKT + kk];
}

// ---------------------------------------------------------------------------
// Direct 3x3 conv, 128 -> 18 channels (offset predictor), pad=1, stride=1.
// Weights staged in LDS (18*128*9 fp32 = 81KB, fits CDNA5 320KB WGP LDS).
// ---------------------------------------------------------------------------
__global__ __launch_bounds__(256) void offset_conv_kernel(
    const float* __restrict__ x, const float* __restrict__ wo,
    const float* __restrict__ bo, float* __restrict__ off) {
    __shared__ float s_w[18 * CC * KKT];
    int tid = threadIdx.x;
    for (int i = tid; i < 18 * CC * KKT; i += 256) s_w[i] = wo[i];
    __syncthreads();

    int b  = blockIdx.x >> 4;
    int rt = blockIdx.x & 15;
    int h  = rt * 4 + (tid >> 6);
    int w  = tid & 63;

    float acc[18];
#pragma unroll
    for (int oc = 0; oc < 18; ++oc) acc[oc] = bo[oc];

    for (int ci = 0; ci < CC; ++ci) {
        const float* xc = x + ((size_t)(b * CC + ci)) * (HH * WW);
        for (int ky = 0; ky < 3; ++ky) {
            int iy = h - 1 + ky;
            if ((unsigned)iy >= HH) continue;
            for (int kx = 0; kx < 3; ++kx) {
                int ix = w - 1 + kx;
                if ((unsigned)ix >= WW) continue;
                float xv = xc[iy * WW + ix];
                const float* wp = &s_w[(ci * 3 + ky) * 3 + kx];
#pragma unroll
                for (int oc = 0; oc < 18; ++oc)
                    acc[oc] = fmaf(xv, wp[oc * KDIM], acc[oc]);
            }
        }
    }
    for (int oc = 0; oc < 18; ++oc)
        off[((size_t)(b * 18 + oc) * HH + h) * WW + w] = acc[oc];
}

// ---------------------------------------------------------------------------
// Deformable conv (K=3, dil=2, pad=2) + BN (+residual) + ReLU, via WMMA.
// Block = 256 threads (8 waves), handles 16 consecutive w-positions of one row.
// ---------------------------------------------------------------------------
__global__ __launch_bounds__(256) void deform_wmma_kernel(
    const float* __restrict__ xsrc,    // sampled tensor (B,128,64,64)
    const float* __restrict__ off,     // offsets (B,18,64,64)
    const _Float16* __restrict__ Bpack,
    const float* __restrict__ g, const float* __restrict__ beta,
    const float* __restrict__ mn, const float* __restrict__ vr,
    const float* __restrict__ resid,   // residual (may be unused)
    float* __restrict__ out, int use_resid) {

    __shared__ _Float16 Alds[16 * AROW];           // 16 x 1152 f16 patch matrix
    __shared__ int   s_iy0[144], s_iy1[144], s_ix0[144], s_ix1[144];
    __shared__ float s_w00[144], s_w01[144], s_w10[144], s_w11[144];

    int tid = threadIdx.x;
    int b   = blockIdx.x >> 8;           // 256 tiles per image
    int til = blockIdx.x & 255;
    int oh  = til >> 2;                  // row
    int w0  = (til & 3) * 16;            // 16-wide column tile

    // Phase 0: per (position m, tap k) bilinear geometry; validity folded into weights.
    if (tid < 144) {
        int m  = tid / 9;
        int k  = tid % 9;
        int ky = k / 3, kx = k % 3;
        int ow = w0 + m;
        size_t obase = ((size_t)b * 18) * (HH * WW) + (size_t)oh * WW + ow;
        float dy = off[obase + (size_t)(2 * k) * (HH * WW)];
        float dx = off[obase + (size_t)(2 * k + 1) * (HH * WW)];
        float ys = (float)(oh - 2 + ky * 2) + dy;
        float xs = (float)(ow - 2 + kx * 2) + dx;
        float y0f = floorf(ys), x0f = floorf(xs);
        float fy = ys - y0f, fx = xs - x0f;
        int iy0 = (int)y0f, ix0 = (int)x0f;
        int iy1 = iy0 + 1, ix1 = ix0 + 1;
        float vy0 = ((unsigned)iy0 < HH) ? 1.f : 0.f;
        float vy1 = ((unsigned)iy1 < HH) ? 1.f : 0.f;
        float vx0 = ((unsigned)ix0 < WW) ? 1.f : 0.f;
        float vx1 = ((unsigned)ix1 < WW) ? 1.f : 0.f;
        s_w00[tid] = (1.f - fy) * (1.f - fx) * vy0 * vx0;
        s_w01[tid] = (1.f - fy) * fx         * vy0 * vx1;
        s_w10[tid] = fy * (1.f - fx)         * vy1 * vx0;
        s_w11[tid] = fy * fx                 * vy1 * vx1;
        s_iy0[tid] = min(max(iy0, 0), HH - 1);
        s_iy1[tid] = min(max(iy1, 0), HH - 1);
        s_ix0[tid] = min(max(ix0, 0), WW - 1);
        s_ix1[tid] = min(max(ix1, 0), WW - 1);
    }
    __syncthreads();

    // Phase 1: gather + bilinear interpolate 16x1152 patch -> LDS (f16).
    for (int s = tid; s < 144 * CC; s += 256) {
        int pk = s >> 7;          // 0..143 = m*9+k
        int ci = s & 127;
        int m  = pk / 9;
        int k  = pk % 9;
        const float* base = xsrc + ((size_t)(b * CC + ci)) * (HH * WW);
        int iy0 = s_iy0[pk], iy1 = s_iy1[pk], ix0 = s_ix0[pk], ix1 = s_ix1[pk];
        float v00 = base[iy0 * WW + ix0];
        float v01 = base[iy0 * WW + ix1];
        float v10 = base[iy1 * WW + ix0];
        float v11 = base[iy1 * WW + ix1];
        float sm = s_w00[pk] * v00 + s_w01[pk] * v01 + s_w10[pk] * v10 + s_w11[pk] * v11;
        Alds[m * AROW + ci * KKT + k] = (_Float16)sm;   // K-index = ci*9 + k
    }
    __syncthreads();

    // Phase 2: 8 waves, each owns a 16-channel N-tile; K = 1152 in 36 x 32 chunks.
    int wave = tid >> 5;
    int lane = tid & 31;
    int hi   = lane >> 4;
    int mrow = lane & 15;        // A-matrix M index for this lane
    int nt   = wave;

    v8f acc = {};
    const uint32_t* arow = (const uint32_t*)(&Alds[mrow * AROW]);  // AROW even -> aligned

    for (int kc = 0; kc < NKC; ++kc) {
        // A fragment per ISA 16-bit 16x32 layout: VGPR j holds (K, K+1);
        // lanes 0-15: K = 2j (j<4) / 16+2(j-4) ; lanes 16-31: +8.
        union { uint32_t u[8]; v16h h; } A;
        int kb2 = kc * 16 + hi * 4;   // (kc*32 + hi*8) / 2, in uint units
#pragma unroll
        for (int j = 0; j < 4; ++j) A.u[j] = arow[kb2 + j];
#pragma unroll
        for (int j = 4; j < 8; ++j) A.u[j] = arow[kb2 + 8 + (j - 4)];

        // B fragment: 32 contiguous bytes per lane from the packed buffer.
        union { uint4 q[2]; v16h h; } Bf;
        const uint4* bp = (const uint4*)(Bpack + ((((size_t)nt * NKC + kc) * 32 + lane) << 4));
        Bf.q[0] = bp[0];
        Bf.q[1] = bp[1];

        acc = __builtin_amdgcn_wmma_f32_16x16x32_f16(
            false, A.h, false, Bf.h, (short)0, acc, false, false);
    }

    // Epilogue: BN + optional residual + ReLU from C/D layout
    // (VGPR r, lane L: M = r + 8*(L>=16), N = L&15).
    int co = nt * 16 + mrow;
    float inv  = g[co] * rsqrtf(vr[co] + 1e-5f);
    float bias = beta[co] - mn[co] * inv;
    size_t cbase = ((size_t)(b * CC + co) * HH + oh) * WW;
#pragma unroll
    for (int r = 0; r < 8; ++r) {
        int mpos = r + 8 * hi;
        int ow = w0 + mpos;
        float val = acc[r] * inv + bias;
        if (use_resid) val += resid[cbase + ow];
        out[cbase + ow] = fmaxf(val, 0.f);
    }
}

// ---------------------------------------------------------------------------
// Launch: pack weights, then stage1 (offset conv -> deform/BN/ReLU),
// then stage2 (offset conv -> deform/BN + residual + ReLU) into d_out.
// ---------------------------------------------------------------------------
extern "C" void kernel_launch(void* const* d_in, const int* in_sizes, int n_in,
                              void* d_out, int out_size, void* d_ws, size_t ws_size,
                              hipStream_t stream) {
    const float* x      = (const float*)d_in[0];
    const float* w_off1 = (const float*)d_in[1];
    const float* b_off1 = (const float*)d_in[2];
    const float* w_dc1  = (const float*)d_in[3];
    const float* g1     = (const float*)d_in[4];
    const float* beta1  = (const float*)d_in[5];
    const float* m1     = (const float*)d_in[6];
    const float* v1     = (const float*)d_in[7];
    const float* w_off2 = (const float*)d_in[8];
    const float* b_off2 = (const float*)d_in[9];
    const float* w_dc2  = (const float*)d_in[10];
    const float* g2     = (const float*)d_in[11];
    const float* beta2  = (const float*)d_in[12];
    const float* m2     = (const float*)d_in[13];
    const float* v2     = (const float*)d_in[14];

    const size_t OFF_BYTES = (size_t)BB * 18 * HH * WW * 4;   // 2,359,296
    const size_t OUT_BYTES = (size_t)BB * CC * HH * WW * 4;   // 16,777,216
    const size_t BP_ELEMS  = (size_t)8 * NKC * 32 * 16;       // 147,456 halves

    char* ws = (char*)d_ws;
    float*     off1 = (float*)ws;
    float*     out1 = (float*)(ws + OFF_BYTES);
    float*     off2 = (float*)(ws + OFF_BYTES + OUT_BYTES);
    _Float16*  bp1  = (_Float16*)(ws + 2 * OFF_BYTES + OUT_BYTES);
    _Float16*  bp2  = bp1 + BP_ELEMS;

    int packBlocks = (int)((BP_ELEMS + 255) / 256);
    pack_w_kernel<<<packBlocks, 256, 0, stream>>>(w_dc1, bp1);
    pack_w_kernel<<<packBlocks, 256, 0, stream>>>(w_dc2, bp2);

    // Stage 1
    offset_conv_kernel<<<BB * 16, 256, 0, stream>>>(x, w_off1, b_off1, off1);
    deform_wmma_kernel<<<BB * 256, 256, 0, stream>>>(
        x, off1, bp1, g1, beta1, m1, v1, x, out1, 0);

    // Stage 2 (residual = x, final ReLU) -> d_out
    offset_conv_kernel<<<BB * 16, 256, 0, stream>>>(out1, w_off2, b_off2, off2);
    deform_wmma_kernel<<<BB * 256, 256, 0, stream>>>(
        out1, off2, bp2, g2, beta2, m2, v2, x, (float*)d_out, 1);
}
